// KarmaSTDP_23330262352057
// MI455X (gfx1250) — matrix operand order repaired and checked by reading, not compiled, not run on hardware
//
#include <hip/hip_runtime.h>

#define T_DIM 8192
#define NDIM  1024
#define DECAY 0.8f
#define POS_K 0.05f
#define NEG_K 0.01f
#define NCH   64     // chunks along T
#define CHLEN 128    // chunk length
#define HALO  64     // 0.8^64 ~ 6e-7 << bf16 eps -> carry-free chunked scan
#define LDSK  40     // padded LDS row stride (halfs): 20 words -> bank spread
#define KSTEP 32
#define NIT   (T_DIM / KSTEP)

typedef __attribute__((ext_vector_type(16))) __bf16        v16bf;
typedef __attribute__((ext_vector_type(8)))  float         v8f;
typedef __attribute__((ext_vector_type(4)))  unsigned int  u32x4;

__device__ __forceinline__ unsigned short f32_bf16(float f) {
    unsigned u = __builtin_bit_cast(unsigned, f);
    u += 0x7FFFu + ((u >> 16) & 1u);   // round-to-nearest-even
    return (unsigned short)(u >> 16);
}

// ---------------------------------------------------------------------------
// Pass A: leaky traces via halo-chunked scan + bf16 conversion with the
// karma scale factors folded into the A-side operands.
//   tib = bf16(0.05 * leaky(x))   xbs = bf16(-0.01 * x)
//   tob = bf16(       leaky(y))   yb  = bf16(        y)
// ---------------------------------------------------------------------------
__global__ __launch_bounds__(256)
void karma_scan(const float* __restrict__ x, const float* __restrict__ y,
                unsigned short* __restrict__ tib, unsigned short* __restrict__ xbs,
                unsigned short* __restrict__ yb,  unsigned short* __restrict__ tob)
{
    const int g = blockIdx.x * blockDim.x + threadIdx.x;   // [0, 2*NCH*NDIM)
    const int j = g & (NDIM - 1);                          // column (lane-contiguous)
    const int c = (g >> 10) & (NCH - 1);                   // chunk
    const int m = g >> 16;                                 // 0:x  1:y

    const float* __restrict__ src = m ? y : x;
    unsigned short* __restrict__ trd = m ? tob : tib;
    unsigned short* __restrict__ cpd = m ? yb  : xbs;
    const float tscale = m ? 1.0f : POS_K;
    const float cscale = m ? 1.0f : -NEG_K;

    const int t0 = c * CHLEN;
    float run = 0.0f;
    #pragma unroll 4
    for (int k = -HALO; k < CHLEN; ++k) {
        const int t = t0 + k;
        if (t < 0) continue;                 // only chunk 0 lacks a halo
        const float v = src[t * NDIM + j];
        run = run * DECAY + v;
        if (k >= 0) {
            trd[t * NDIM + j] = f32_bf16(run * tscale);
            cpd[t * NDIM + j] = f32_bf16(v * cscale);
        }
    }
}

// ---------------------------------------------------------------------------
// Pass B: out[i,o] = samskaras[i,o] + tib^T*yb + xbs^T*tob
// 128x128 block tile, 8 waves (4x2) of 32x64, K-step 32, bf16 WMMA, f32 acc.
// Double-buffered LDS (80 KB dynamic), one barrier per K-step.
// ---------------------------------------------------------------------------
__global__ __launch_bounds__(256)
void karma_gemm(const unsigned short* __restrict__ tib,
                const unsigned short* __restrict__ xbs,
                const unsigned short* __restrict__ yb,
                const unsigned short* __restrict__ tob,
                const float* __restrict__ samskaras,
                float* __restrict__ out)
{
    extern __shared__ unsigned short smem[];   // [2 bufs][4 mats][128 rows][LDSK]

    const int tid   = threadIdx.x;
    const int lane  = tid & 31;
    const int w     = tid >> 5;
    const int laneq = lane & 15;
    const int laneh = lane >> 4;
    const int i0  = blockIdx.x * 128;
    const int o0  = blockIdx.y * 128;
    const int wm0 = (w & 3) * 32;
    const int wn0 = (w >> 2) * 64;

    // staging: thread owns a row PAIR (u32 = 2 bf16 rows -> 128B coalesced
    // wave transactions) and an 8-wide k quarter.
    const int rp = (tid & 63) << 1;            // 0,2,...,126
    const int kq = (tid >> 6) << 3;            // 0,8,16,24
    const unsigned* __restrict__ gsrc[4] = {
        (const unsigned*)(tib + i0 + rp), (const unsigned*)(xbs + i0 + rp),
        (const unsigned*)(yb  + o0 + rp), (const unsigned*)(tob + o0 + rp) };

    auto stage = [&](int kt, int buf) {
        #pragma unroll
        for (int mat = 0; mat < 4; ++mat) {
            union { unsigned short s[8]; u32x4 q; } lo, hi;
            #pragma unroll
            for (int kk = 0; kk < 8; ++kk) {
                const unsigned v = gsrc[mat][(size_t)(kt + kq + kk) * (NDIM / 2)];
                lo.s[kk] = (unsigned short)(v & 0xFFFFu);
                hi.s[kk] = (unsigned short)(v >> 16);
            }
            unsigned short* dst = smem + (((buf * 4 + mat) * 128 + rp) * LDSK + kq);
            *(u32x4*)dst          = lo.q;      // row rp
            *(u32x4*)(dst + LDSK) = hi.q;      // row rp+1
        }
    };

    v8f acc[2][4];
    #pragma unroll
    for (int mt = 0; mt < 2; ++mt)
        #pragma unroll
        for (int nt = 0; nt < 4; ++nt)
            acc[mt][nt] = (v8f){0.f,0.f,0.f,0.f,0.f,0.f,0.f,0.f};

    stage(0, 0);
    #pragma unroll 2
    for (int it = 0; it < NIT; ++it) {
        __syncthreads();                       // buf `it&1` fully staged
        const int buf = it & 1;

        // issue next tile's global loads early; lands in the other buffer
        if (it + 1 < NIT) stage((it + 1) * KSTEP, buf ^ 1);

        // ---- load fragments from current buffer ----
        // A (16x32 bf16): lane L -> row M=L%16; K chunks {8h..+7, 8h+16..+23}
        v16bf fa[2][2];
        #pragma unroll
        for (int mat = 0; mat < 2; ++mat)
            #pragma unroll
            for (int mt = 0; mt < 2; ++mt) {
                const unsigned short* p = smem +
                    (((buf * 4 + mat) * 128 + wm0 + mt * 16 + laneq) * LDSK + laneh * 8);
                union { u32x4 q[2]; v16bf v; } u;
                u.q[0] = *(const u32x4*)p;
                u.q[1] = *(const u32x4*)(p + 16);
                fa[mat][mt] = u.v;
            }
        // B (32x16 bf16): lane L -> col N=L%16; K chunks {8h..+7, 8h+16..+23}
        v16bf fb[2][4];
        #pragma unroll
        for (int mat = 0; mat < 2; ++mat)
            #pragma unroll
            for (int nt = 0; nt < 4; ++nt) {
                const unsigned short* p = smem +
                    (((buf * 4 + 2 + mat) * 128 + wn0 + nt * 16 + laneq) * LDSK + laneh * 8);
                union { u32x4 q[2]; v16bf v; } u;
                u.q[0] = *(const u32x4*)p;
                u.q[1] = *(const u32x4*)(p + 16);
                fb[mat][nt] = u.v;
            }

        // ---- 16 WMMAs: both fused GEMMs into one accumulator set ----
        #pragma unroll
        for (int mt = 0; mt < 2; ++mt)
            #pragma unroll
            for (int nt = 0; nt < 4; ++nt) {
                acc[mt][nt] = __builtin_amdgcn_wmma_f32_16x16x32_bf16(
                    false, fa[0][mt], false, fb[0][nt], (short)0, acc[mt][nt], false, false);
                acc[mt][nt] = __builtin_amdgcn_wmma_f32_16x16x32_bf16(
                    false, fa[1][mt], false, fb[1][nt], (short)0, acc[mt][nt], false, false);
            }
    }

    // ---- epilogue: C/D layout M = v + 8*laneh, N = laneq ----
    #pragma unroll
    for (int mt = 0; mt < 2; ++mt)
        #pragma unroll
        for (int nt = 0; nt < 4; ++nt)
            #pragma unroll
            for (int v = 0; v < 8; ++v) {
                const int i = i0 + wm0 + mt * 16 + laneh * 8 + v;
                const int o = o0 + wn0 + nt * 16 + laneq;
                const size_t idx = (size_t)i * NDIM + o;
                out[idx] = samskaras[idx] + acc[mt][nt][v];
            }
}

// ---------------------------------------------------------------------------
extern "C" void kernel_launch(void* const* d_in, const int* in_sizes, int n_in,
                              void* d_out, int out_size, void* d_ws, size_t ws_size,
                              hipStream_t stream)
{
    const float* x         = (const float*)d_in[0];   // prana_in  (8192,1,1024) f32
    const float* y         = (const float*)d_in[1];   // buddhi_out(8192,1,1024) f32
    const float* samskaras = (const float*)d_in[2];   // (1024,1024) f32
    float* out             = (float*)d_out;

    const size_t matN = (size_t)T_DIM * NDIM;         // elements per operand matrix
    unsigned short* tib = (unsigned short*)d_ws;      // 4 x 16 MB bf16 in workspace
    unsigned short* xbs = tib + matN;
    unsigned short* yb  = xbs + matN;
    unsigned short* tob = yb  + matN;

    karma_scan<<<(2 * NCH * NDIM) / 256, 256, 0, stream>>>(x, y, tib, xbs, yb, tob);

    const size_t smemBytes = 2u * 4u * 128u * LDSK * sizeof(unsigned short); // 80 KB
    karma_gemm<<<dim3(NDIM / 128, NDIM / 128), 256, smemBytes, stream>>>(
        tib, xbs, yb, tob, samskaras, out);
}
